// DGCNLayer_9070970929776
// MI455X (gfx1250) — compile-verified, matrix operand order checked by reading, not run on hardware
//
#include <hip/hip_runtime.h>
#include <hip/hip_bf16.h>
#include <math.h>

#define D 128
#define R 4

typedef __attribute__((ext_vector_type(16))) __bf16 v16bf;
typedef __attribute__((ext_vector_type(8)))  __bf16 v8bf;
typedef __attribute__((ext_vector_type(8)))  float  v8f;

// ---------- bit-exact RNE float->bf16 (prep path only) ----------
__device__ __forceinline__ unsigned short bfbits(float f) {
    union { float f; unsigned int u; } v; v.f = f;
    unsigned int r = v.u + 0x7FFFu + ((v.u >> 16) & 1u);
    return (unsigned short)(r >> 16);
}

// ---------- zero fill ----------
__global__ void zero_kernel(float4* __restrict__ p, long n4) {
    long i = (long)blockIdx.x * blockDim.x + threadIdx.x;
    if (i < n4) p[i] = make_float4(0.f, 0.f, 0.f, 0.f);
}

// ---------- softmax(rel_weights) + c_r = rel_emb @ W_fuse[D:] + b_fuse ----------
__global__ void prep_small(const float* __restrict__ rw, const float* __restrict__ relE,
                           const float* __restrict__ W_fuse, const float* __restrict__ b_fuse,
                           float* __restrict__ wsoft, float* __restrict__ crel) {
    int tid = threadIdx.x;
    float m = fmaxf(fmaxf(rw[0], rw[1]), fmaxf(rw[2], rw[3]));
    float s = expf(rw[0] - m) + expf(rw[1] - m) + expf(rw[2] - m) + expf(rw[3] - m);
    if (tid < R) wsoft[tid] = expf(rw[tid] - m) / s;
    for (int e = tid; e < R * D; e += blockDim.x) {
        int r = e >> 7, d = e & 127;
        float acc = b_fuse[d];
        for (int k = 0; k < D; ++k)
            acc += relE[r * D + k] * W_fuse[(D + k) * D + d];
        crel[e] = acc;
    }
}

// ---------- convert weights to bf16 in FRAGMENT-MAJOR (swizzled) layout ----------
// consumed as: frag f = kb*8+t; element j of lane l lives at ((f*32+l)*16+j)
// value = W[k][n], n = t*16 + (l&15), hf = l>>4,
//         k = kb*32 + (j<8 ? hf*8+j : 16+hf*8+(j-8))
__global__ void prep_weights(const float* __restrict__ W_fuse, const float* __restrict__ W_rel,
                             const float* __restrict__ W_gate,
                             unsigned short* __restrict__ owf, unsigned short* __restrict__ owr,
                             unsigned short* __restrict__ owg) {
    int i = blockIdx.x * 256 + threadIdx.x;   // 0 .. 98303  (6 matrices x 16384)
    int mat = i >> 14;                         // 0:Wf_top, 1..4:W_rel[r], 5:W_gate
    int idx = i & 16383;
    int f  = idx >> 9;
    int l  = (idx >> 4) & 31;
    int j  = idx & 15;
    int kb = f >> 3, t = f & 7;
    int hf = l >> 4;
    int n  = t * 16 + (l & 15);
    int k  = kb * 32 + ((j < 8) ? (hf * 8 + j) : (16 + hf * 8 + (j - 8)));
    int src = k * D + n;
    if (mat == 0)      owf[idx] = bfbits(W_fuse[src]);                       // top 128 rows
    else if (mat <= 4) owr[(mat - 1) * 16384 + idx] = bfbits(W_rel[(mat - 1) * 16384 + src]);
    else               owg[idx] = bfbits(W_gate[src]);
}

// ---------- SpMM scatter: msg[rows[e]] += vals[e] * X[cols[e]] ----------
__global__ void spmm_kernel(const float* __restrict__ X, const int* __restrict__ rows,
                            const int* __restrict__ cols, const float* __restrict__ vals,
                            float* __restrict__ msg, int E) {
    int e = blockIdx.x * 8 + (threadIdx.x >> 5);
    if (e >= E) return;
    int lane = threadIdx.x & 31;
    int c = cols[e], rdst = rows[e];
    float v = vals[e];
    float4 x4 = ((const float4*)(X + (long)c * D))[lane];
    float* m = msg + (long)rdst * D + lane * 4;
    atomicAdd(m + 0, v * x4.x);
    atomicAdd(m + 1, v * x4.y);
    atomicAdd(m + 2, v * x4.z);
    atomicAdd(m + 3, v * x4.w);
}

// ---------- build A fragment from 16 fp32 values (wide b128 loads + native cvt) ----------
__device__ __forceinline__ v16bf load_a_f32(const float* __restrict__ p) {
    // p points at row base + k0 + hf*8 ; runs [0..7] and [16..23]
    float4 a0 = *(const float4*)(p);
    float4 a1 = *(const float4*)(p + 4);
    float4 a2 = *(const float4*)(p + 16);
    float4 a3 = *(const float4*)(p + 20);
    v16bf a;
    a[0] = (__bf16)a0.x;  a[1] = (__bf16)a0.y;  a[2] = (__bf16)a0.z;  a[3] = (__bf16)a0.w;
    a[4] = (__bf16)a1.x;  a[5] = (__bf16)a1.y;  a[6] = (__bf16)a1.z;  a[7] = (__bf16)a1.w;
    a[8] = (__bf16)a2.x;  a[9] = (__bf16)a2.y;  a[10] = (__bf16)a2.z; a[11] = (__bf16)a2.w;
    a[12] = (__bf16)a3.x; a[13] = (__bf16)a3.y; a[14] = (__bf16)a3.z; a[15] = (__bf16)a3.w;
    return a;
}

// ---------- per-relation fused double GEMM, 128 threads (4 waves), 64 rows/block ----------
__global__ __launch_bounds__(128) void gemm_rel_kernel(
    const float* __restrict__ msg, const unsigned short* __restrict__ WfB,
    const unsigned short* __restrict__ WrelB, const float* __restrict__ crel,
    const float* __restrict__ brel, const float* __restrict__ wsoft,
    float* __restrict__ combined, int r, int N) {
    __shared__ __attribute__((aligned(32))) __bf16 Wlds[D * D];   // 32 KB, fragment-major
    __shared__ __attribute__((aligned(32))) __bf16 Fstg[64 * D];  // 16 KB, row-major

    const int tid  = threadIdx.x;
    const int wave = tid >> 5;
    const int lane = tid & 31;
    const int hf   = lane >> 4;
    const int ln16 = lane & 15;
    const long row0 = (long)blockIdx.x * 64;
    const long rw0  = row0 + wave * 16;
    const bool active = rw0 < N;                 // wave-uniform (N % 16 == 0)

    { // wide weight copy: Wf_top
        const uint4* s = (const uint4*)WfB; uint4* d = (uint4*)Wlds;
        for (int i = tid; i < D * D / 8; i += 128) d[i] = s[i];
    }
    __syncthreads();

    long mrow = rw0 + ln16; if (mrow >= N) mrow = N - 1;   // clamp tail loads
    const float* mp = msg + mrow * D;

    v8f acc[8] = {};
    #pragma unroll
    for (int kb = 0; kb < 4; ++kb) {
        v16bf a = load_a_f32(mp + kb * 32 + hf * 8);
        #pragma unroll
        for (int t = 0; t < 8; ++t) {
            v16bf b = *(const v16bf*)(Wlds + (((kb * 8 + t) * 32 + lane) << 4));
            acc[t] = __builtin_amdgcn_wmma_f32_16x16x32_bf16(
                false, a, false, b, (short)0, acc[t], false, false);
        }
    }

    // bias + relu, restage row-major bf16
    #pragma unroll
    for (int t = 0; t < 8; ++t) {
        int n = t * 16 + ln16;
        float c = crel[r * D + n];
        #pragma unroll
        for (int i = 0; i < 8; ++i) {
            float f = acc[t][i] + c;
            f = f > 0.f ? f : 0.f;
            Fstg[(wave * 16 + i + 8 * hf) * D + n] = (__bf16)f;
        }
    }
    __syncthreads();
    { // wide weight copy: W_rel[r]
        const uint4* s = (const uint4*)(WrelB + r * D * D); uint4* d = (uint4*)Wlds;
        for (int i = tid; i < D * D / 8; i += 128) d[i] = s[i];
    }
    __syncthreads();

    v8f acc2[8] = {};
    const __bf16* fp = Fstg + (wave * 16 + ln16) * D;
    #pragma unroll
    for (int kb = 0; kb < 4; ++kb) {
        const int k0 = kb * 32;
        v8bf lo = *(const v8bf*)(fp + k0 + hf * 8);
        v8bf hi = *(const v8bf*)(fp + k0 + 16 + hf * 8);
        v16bf a = __builtin_shufflevector(lo, hi, 0,1,2,3,4,5,6,7,8,9,10,11,12,13,14,15);
        #pragma unroll
        for (int t = 0; t < 8; ++t) {
            v16bf b = *(const v16bf*)(Wlds + (((kb * 8 + t) * 32 + lane) << 4));
            acc2[t] = __builtin_amdgcn_wmma_f32_16x16x32_bf16(
                false, a, false, b, (short)0, acc2[t], false, false);
        }
    }

    if (active) {
        const float w = wsoft[r];
        #pragma unroll
        for (int t = 0; t < 8; ++t) {
            int n = t * 16 + ln16;
            float bb = brel[r * D + n];
            #pragma unroll
            for (int i = 0; i < 8; ++i)
                combined[(rw0 + i + 8 * hf) * D + n] += w * (acc2[t][i] + bb);
        }
    }
}

// ---------- gate GEMM + residual:  comb <- X + sigmoid(X@Wg + bg) * comb ----------
__global__ __launch_bounds__(128) void gate_kernel(
    const float* __restrict__ X, const unsigned short* __restrict__ WgB,
    const float* __restrict__ bg, float* __restrict__ comb, int N) {
    __shared__ __attribute__((aligned(32))) __bf16 Wlds[D * D];
    const int tid  = threadIdx.x;
    const int wave = tid >> 5;
    const int lane = tid & 31;
    const int hf   = lane >> 4;
    const int ln16 = lane & 15;
    const long rw0 = (long)blockIdx.x * 64 + wave * 16;
    const bool active = rw0 < N;

    { const uint4* s = (const uint4*)WgB; uint4* d = (uint4*)Wlds;
      for (int i = tid; i < D * D / 8; i += 128) d[i] = s[i]; }
    __syncthreads();

    long mrow = rw0 + ln16; if (mrow >= N) mrow = N - 1;
    const float* xp = X + mrow * D;

    v8f acc[8] = {};
    #pragma unroll
    for (int kb = 0; kb < 4; ++kb) {
        v16bf a = load_a_f32(xp + kb * 32 + hf * 8);
        #pragma unroll
        for (int t = 0; t < 8; ++t) {
            v16bf b = *(const v16bf*)(Wlds + (((kb * 8 + t) * 32 + lane) << 4));
            acc[t] = __builtin_amdgcn_wmma_f32_16x16x32_bf16(
                false, a, false, b, (short)0, acc[t], false, false);
        }
    }

    if (active) {
        #pragma unroll
        for (int t = 0; t < 8; ++t) {
            int n = t * 16 + ln16;
            float bgn = bg[n];
            #pragma unroll
            for (int i = 0; i < 8; ++i) {
                long idx = (rw0 + i + 8 * hf) * D + n;
                float g = 1.f / (1.f + expf(-(acc[t][i] + bgn)));
                comb[idx] = X[idx] + g * comb[idx];
            }
        }
    }
}

// ---------- LayerNorm, one wave per row, in-place ----------
__global__ __launch_bounds__(256) void ln_kernel(float* __restrict__ out,
                                                 const float* __restrict__ gamma,
                                                 const float* __restrict__ beta) {
    long row = (long)blockIdx.x * 8 + (threadIdx.x >> 5);
    int lane = threadIdx.x & 31;
    float4 xv = *(const float4*)(out + row * D + lane * 4);

    float s = xv.x + xv.y + xv.z + xv.w;
    #pragma unroll
    for (int off = 16; off > 0; off >>= 1) s += __shfl_xor(s, off, 32);
    float mu = s * (1.f / 128.f);

    float d0 = xv.x - mu, d1 = xv.y - mu, d2 = xv.z - mu, d3 = xv.w - mu;
    float q = d0 * d0 + d1 * d1 + d2 * d2 + d3 * d3;
    #pragma unroll
    for (int off = 16; off > 0; off >>= 1) q += __shfl_xor(q, off, 32);
    float inv = rsqrtf(q * (1.f / 128.f) + 1e-3f);

    int n = lane * 4;
    float4 o;
    o.x = d0 * inv * gamma[n + 0] + beta[n + 0];
    o.y = d1 * inv * gamma[n + 1] + beta[n + 1];
    o.z = d2 * inv * gamma[n + 2] + beta[n + 2];
    o.w = d3 * inv * gamma[n + 3] + beta[n + 3];
    *(float4*)(out + row * D + lane * 4) = o;
}

extern "C" void kernel_launch(void* const* d_in, const int* in_sizes, int n_in,
                              void* d_out, int out_size, void* d_ws, size_t ws_size,
                              hipStream_t stream) {
    const float* X      = (const float*)d_in[0];
    const float* relE   = (const float*)d_in[1];
    const int*   rows   = (const int*)d_in[2];
    const int*   cols   = (const int*)d_in[3];
    const float* vals   = (const float*)d_in[4];
    const float* W_fuse = (const float*)d_in[5];
    const float* b_fuse = (const float*)d_in[6];
    const float* W_rel  = (const float*)d_in[7];
    const float* b_rel  = (const float*)d_in[8];
    const float* rw     = (const float*)d_in[9];
    const float* W_gate = (const float*)d_in[10];
    const float* b_gate = (const float*)d_in[11];
    const float* gamma  = (const float*)d_in[12];
    const float* beta   = (const float*)d_in[13];
    float* out = (float*)d_out;

    const int Nn = in_sizes[0] / D;        // 100000
    const int E  = in_sizes[2] / R;        // 1600000

    // workspace layout
    float* msg   = (float*)d_ws;                       // N*D floats
    float* wsoft = msg + (long)Nn * D;                 // 4
    float* crel  = wsoft + R;                          // R*D
    unsigned short* WfB   = (unsigned short*)(crel + R * D);
    unsigned short* WrelB = WfB + D * D;
    unsigned short* WgB   = WrelB + R * D * D;

    prep_small<<<1, 128, 0, stream>>>(rw, relE, W_fuse, b_fuse, wsoft, crel);
    prep_weights<<<384, 256, 0, stream>>>(W_fuse, W_rel, W_gate, WfB, WrelB, WgB);

    const long n4 = (long)Nn * D / 4;
    zero_kernel<<<(int)((n4 + 255) / 256), 256, 0, stream>>>((float4*)out, n4);

    const int gblocks = (Nn + 63) / 64;
    for (int r = 0; r < R; ++r) {
        zero_kernel<<<(int)((n4 + 255) / 256), 256, 0, stream>>>((float4*)msg, n4);
        spmm_kernel<<<(E + 7) / 8, 256, 0, stream>>>(
            X, rows + (long)r * E, cols + (long)r * E, vals + (long)r * E, msg, E);
        gemm_rel_kernel<<<gblocks, 128, 0, stream>>>(
            msg, WfB, WrelB, crel, b_rel, wsoft, out, r, Nn);
    }

    gate_kernel<<<gblocks, 128, 0, stream>>>(X, WgB, b_gate, out, Nn);
    ln_kernel<<<Nn / 8, 256, 0, stream>>>(out, gamma, beta);
}
